// GraphAttentionEmbedding_7344394076398
// MI455X (gfx1250) — matrix operand order repaired and checked by reading, not compiled, not run on hardware
//
#include <hip/hip_runtime.h>
#include <hip/hip_bf16.h>
#include <math.h>

// MI455X / gfx1250, wave32. WMMA f16->f32 for all GEMM work; operands
// pre-converted to packed f16 so hot loops are pure b128-load + wmma.
// Primary pipeline: CSR-by-dst, single WMMA edge pass, atomic-free
// aggregation. Fallback (small ws): round-2 atomic pipeline.

typedef __attribute__((ext_vector_type(16))) _Float16 v16h;
typedef __attribute__((ext_vector_type(8)))  _Float16 v8h;
typedef __attribute__((ext_vector_type(4)))  _Float16 v4h;
typedef __attribute__((ext_vector_type(8)))  float    v8f;

#define N_NODES  50000
#define N_EDGES  500000
#define D_IN     128
#define EDGE_DIM 256
#define HC       128   // heads*d_head = 2*64

__device__ __forceinline__ v16h mk16(v8h lo, v8h hi8) {
  v16h r;
#pragma unroll
  for (int j = 0; j < 8; ++j) { r[j] = lo[j]; r[8 + j] = hi8[j]; }
  return r;
}

// ---- ordered-uint encoding for float atomic max (fallback path) ----
__device__ __forceinline__ unsigned enc_f32(float f) {
  unsigned u = __float_as_uint(f);
  return (u & 0x80000000u) ? ~u : (u | 0x80000000u);
}
__device__ __forceinline__ float dec_f32(unsigned e) {
  return (e & 0x80000000u) ? __uint_as_float(e & 0x7FFFFFFFu)
                           : __uint_as_float(~e);
}

// ------------------------------------------------------------------
// Prep kernels
// ------------------------------------------------------------------
__global__ void k_to_f16(const float* __restrict__ s, _Float16* __restrict__ d, int n) {
  int i = blockIdx.x * 256 + threadIdx.x;
  if (i < n) d[i] = (_Float16)s[i];
}

__global__ void k_zero_int(int* __restrict__ p, int n) {
  int i = blockIdx.x * 256 + threadIdx.x;
  if (i < n) p[i] = 0;
}

__global__ void k_init(unsigned* __restrict__ max_ws, float* __restrict__ den_ws) {
  int i = blockIdx.x * 256 + threadIdx.x;
  if (i < N_NODES * 2) { max_ws[i] = 0x007FFFFFu; den_ws[i] = 0.0f; }
}

// ------------------------------------------------------------------
// K1: node projections q/k/v/skip (shared by both pipelines).
// ------------------------------------------------------------------
__global__ __launch_bounds__(256)
void k_node_proj(const _Float16* __restrict__ xh,
                 const _Float16* __restrict__ wh,          // [4][128][128] f16
                 const float* __restrict__ bq, const float* __restrict__ bk,
                 const float* __restrict__ bv, const float* __restrict__ bs,
                 float* __restrict__ qo, float* __restrict__ ko,
                 float* __restrict__ vo, float* __restrict__ so) {
  const int lane = threadIdx.x & 31;
  const int wave = threadIdx.x >> 5;
  const int rowBase = blockIdx.x * 16;       // 50000/16 = 3125 exact
  const int mrow = lane & 15;
  const bool hi = (lane & 16) != 0;
  const int col = wave * 16 + mrow;

  const float* B[4] = {bq, bk, bv, bs};
  float*       O[4] = {qo, ko, vo, so};

  v8f acc[4];
#pragma unroll
  for (int mi = 0; mi < 4; ++mi) {
    float bias = B[mi][col];
#pragma unroll
    for (int r = 0; r < 8; ++r) acc[mi][r] = bias;
  }

  const _Float16* xrow = xh + (size_t)(rowBase + mrow) * D_IN;
#pragma unroll
  for (int kt = 0; kt < 4; ++kt) {
    const int akb = kt * 32 + (hi ? 16 : 0);
    const v16h a = mk16(*(const v8h*)(xrow + akb),
                        *(const v8h*)(xrow + akb + 8));
    const int bkb = kt * 32 + (hi ? 16 : 0);
#pragma unroll
    for (int mi = 0; mi < 4; ++mi) {
      const _Float16* wr = wh + (size_t)mi * (D_IN * HC) + (size_t)col * D_IN + bkb;
      const v16h b = mk16(*(const v8h*)wr, *(const v8h*)(wr + 8));
      acc[mi] = __builtin_amdgcn_wmma_f32_16x16x32_f16(
          false, a, false, b, (short)0, acc[mi], false, false);
    }
  }

  const int rbase = rowBase + (hi ? 8 : 0);
#pragma unroll
  for (int mi = 0; mi < 4; ++mi)
#pragma unroll
    for (int r = 0; r < 8; ++r)
      O[mi][(size_t)(rbase + r) * HC + col] = acc[mi][r];
}

// ------------------------------------------------------------------
// CSR construction: degree count, 3-kernel exclusive scan, scatter.
// ------------------------------------------------------------------
__global__ void k_degree(const int* __restrict__ ei, int* __restrict__ deg) {
  int e = blockIdx.x * 256 + threadIdx.x;
  if (e < N_EDGES) atomicAdd(&deg[ei[N_EDGES + e]], 1);
}

__global__ void k_scan1(const int* __restrict__ deg, int* __restrict__ off,
                        int* __restrict__ bsum) {
  __shared__ int s[256];
  int tid = threadIdx.x;
  int i = blockIdx.x * 256 + tid;
  int v = (i < N_NODES) ? deg[i] : 0;
  s[tid] = v; __syncthreads();
  for (int o = 1; o < 256; o <<= 1) {
    int tval = (tid >= o) ? s[tid - o] : 0;
    __syncthreads();
    s[tid] += tval;
    __syncthreads();
  }
  if (i < N_NODES) off[i] = s[tid] - v;           // exclusive within block
  if (tid == 255) bsum[blockIdx.x] = s[255];
}

__global__ void k_scan2(const int* __restrict__ bsum, int* __restrict__ bofs, int nb) {
  __shared__ int s[256];
  int tid = threadIdx.x;
  int v = (tid < nb) ? bsum[tid] : 0;
  s[tid] = v; __syncthreads();
  for (int o = 1; o < 256; o <<= 1) {
    int tval = (tid >= o) ? s[tid - o] : 0;
    __syncthreads();
    s[tid] += tval;
    __syncthreads();
  }
  if (tid < nb) bofs[tid] = s[tid] - v;
}

__global__ void k_scan3(int* __restrict__ off, const int* __restrict__ bofs,
                        int* __restrict__ cursor) {
  int i = blockIdx.x * 256 + threadIdx.x;
  if (i < N_NODES) {
    int v = off[i] + bofs[blockIdx.x];
    off[i] = v;
    cursor[i] = v;
  }
}

__global__ void k_scatter(const int* __restrict__ ei, int* __restrict__ cursor,
                          int* __restrict__ csr) {
  int e = blockIdx.x * 256 + threadIdx.x;
  if (e < N_EDGES) {
    int p = atomicAdd(&cursor[ei[N_EDGES + e]], 1);
    csr[p] = e;
  }
}

// ------------------------------------------------------------------
// CSR edge pass: ONE WMMA pass produces logits AND v = v_node[src]+e_proj
// (stored f16). 32 edges/block; edge_attr staged in LDS as f16.
// ------------------------------------------------------------------
__global__ __launch_bounds__(256)
void k_edge_compute(const float* __restrict__ last_update,
                    const int* __restrict__ ei,
                    const float* __restrict__ t,
                    const float* __restrict__ msg,
                    const float* __restrict__ time_w,
                    const float* __restrict__ time_b,
                    const _Float16* __restrict__ weh,      // [128][256] f16
                    const float* __restrict__ q_ws,
                    const float* __restrict__ k_ws,
                    const float* __restrict__ v_ws,
                    float* __restrict__ logit_ws,
                    _Float16* __restrict__ vh) {
  __shared__ __align__(16) _Float16 s_attr[32][EDGE_DIM];
  __shared__ int   s_src[32], s_dst[32];
  __shared__ float s_relt[32];
  __shared__ float s_logit[32][2];

  const int tid = threadIdx.x;
  const int base = blockIdx.x * 32;          // 500000/32 = 15625 exact

  if (tid < 32) {
    int e = base + tid;
    int s = ei[e];
    s_src[tid]  = s;
    s_dst[tid]  = ei[N_EDGES + e];
    s_relt[tid] = last_update[s] - t[e];
  }
  if (tid < 64) s_logit[tid >> 1][tid & 1] = 0.0f;
  __syncthreads();

  {  // edge_attr = [cos(rel_t*w+b) | msg] in f16
    int c = tid;
    if (c < 128) {
      float tw = time_w[c], tb = time_b[c];
#pragma unroll
      for (int ee = 0; ee < 32; ++ee)
        s_attr[ee][c] = (_Float16)cosf(s_relt[ee] * tw + tb);
    } else {
      int mc = c - 128;
#pragma unroll
      for (int ee = 0; ee < 32; ++ee)
        s_attr[ee][c] = (_Float16)msg[(size_t)(base + ee) * 128 + mc];
    }
  }
  __syncthreads();

  const int lane = tid & 31;
  const int wave = tid >> 5;
  const int mrow = lane & 15;
  const bool hi = (lane & 16) != 0;
  const int col = wave * 16 + mrow;
  const int er0 = mrow, er1 = 16 + mrow;

  v8f acc0 = {}, acc1 = {};
#pragma unroll
  for (int kt = 0; kt < 8; ++kt) {           // K = 256
    const int akb = kt * 32 + (hi ? 16 : 0);
    const int bkb = kt * 32 + (hi ? 16 : 0);
    const _Float16* wr = weh + (size_t)col * EDGE_DIM + bkb;
    const v16h b  = mk16(*(const v8h*)wr, *(const v8h*)(wr + 8));
    const v16h a0 = mk16(*(const v8h*)&s_attr[er0][akb],
                         *(const v8h*)&s_attr[er0][akb + 8]);
    const v16h a1 = mk16(*(const v8h*)&s_attr[er1][akb],
                         *(const v8h*)&s_attr[er1][akb + 8]);
    acc0 = __builtin_amdgcn_wmma_f32_16x16x32_f16(
        false, a0, false, b, (short)0, acc0, false, false);
    acc1 = __builtin_amdgcn_wmma_f32_16x16x32_f16(
        false, a1, false, b, (short)0, acc1, false, false);
  }

  const int h = col >> 6;
#pragma unroll
  for (int r = 0; r < 8; ++r) {
    int ee = r + (hi ? 8 : 0);
    {
      float ep = acc0[r];
      float kv = ep + k_ws[(size_t)s_src[ee] * HC + col];
      float qv = q_ws[(size_t)s_dst[ee] * HC + col];
      atomicAdd(&s_logit[ee][h], kv * qv);                 // ds_add_f32
      vh[(size_t)(base + ee) * HC + col] =
          (_Float16)(ep + v_ws[(size_t)s_src[ee] * HC + col]);
    }
    int e2 = 16 + ee;
    {
      float ep = acc1[r];
      float kv = ep + k_ws[(size_t)s_src[e2] * HC + col];
      float qv = q_ws[(size_t)s_dst[e2] * HC + col];
      atomicAdd(&s_logit[e2][h], kv * qv);
      vh[(size_t)(base + e2) * HC + col] =
          (_Float16)(ep + v_ws[(size_t)s_src[e2] * HC + col]);
    }
  }
  __syncthreads();

  if (tid < 64) {
    int ee = tid >> 1, hh = tid & 1;
    logit_ws[(size_t)(base + ee) * 2 + hh] = s_logit[ee][hh] * 0.125f;
  }
}

// ------------------------------------------------------------------
// CSR aggregation: one wave per destination node; shfl-reduced local
// softmax, lanes own 4 output columns each; zero atomics.
// ------------------------------------------------------------------
__global__ __launch_bounds__(256)
void k_node_agg(const int* __restrict__ off, const int* __restrict__ csr,
                const float* __restrict__ logit_ws,
                const _Float16* __restrict__ vh,
                float* __restrict__ out) {
  const int lane = threadIdx.x & 31;
  const int wave = threadIdx.x >> 5;
  const int node = blockIdx.x * 8 + wave;    // 6250*8 = 50000 exact
  const int beg = off[node];
  const int end = (node + 1 < N_NODES) ? off[node + 1] : N_EDGES;
  const int deg = end - beg;
  if (deg == 0) return;                      // out keeps skip projection

  float m0 = -1e30f, m1 = -1e30f;
  for (int i = lane; i < deg; i += 32) {
    int eid = csr[beg + i];
    m0 = fmaxf(m0, logit_ws[(size_t)eid * 2]);
    m1 = fmaxf(m1, logit_ws[(size_t)eid * 2 + 1]);
  }
#pragma unroll
  for (int o = 16; o; o >>= 1) {
    m0 = fmaxf(m0, __shfl_xor(m0, o, 32));
    m1 = fmaxf(m1, __shfl_xor(m1, o, 32));
  }
  float d0 = 0.0f, d1 = 0.0f;
  for (int i = lane; i < deg; i += 32) {
    int eid = csr[beg + i];
    d0 += __expf(logit_ws[(size_t)eid * 2] - m0);
    d1 += __expf(logit_ws[(size_t)eid * 2 + 1] - m1);
  }
#pragma unroll
  for (int o = 16; o; o >>= 1) {
    d0 += __shfl_xor(d0, o, 32);
    d1 += __shfl_xor(d1, o, 32);
  }

  const int cb = lane * 4;                   // lane owns cols cb..cb+3
  const int hh = (cb >= 64) ? 1 : 0;
  const float m = hh ? m1 : m0;
  const float inv = 1.0f / (hh ? d1 : d0);
  float acc[4] = {0.f, 0.f, 0.f, 0.f};
  for (int j = 0; j < deg; ++j) {
    int eid = csr[beg + j];
    float alpha = __expf(logit_ws[(size_t)eid * 2 + hh] - m) * inv;
    v4h vr = *(const v4h*)(vh + (size_t)eid * HC + cb);   // 8B coalesced
#pragma unroll
    for (int c = 0; c < 4; ++c) acc[c] += alpha * (float)vr[c];
  }
  float* op = out + (size_t)node * HC + cb;
#pragma unroll
  for (int c = 0; c < 4; ++c) op[c] += acc[c];
}

// ==================================================================
// Fallback pipeline (round-2): used only if ws too small for CSR.
// ==================================================================
__global__ __launch_bounds__(256)
void k_edge_logits(const float* __restrict__ last_update,
                   const int* __restrict__ ei,
                   const float* __restrict__ t,
                   const float* __restrict__ msg,
                   const float* __restrict__ time_w,
                   const float* __restrict__ time_b,
                   const _Float16* __restrict__ weh,
                   const float* __restrict__ q_ws,
                   const float* __restrict__ k_ws,
                   float* __restrict__ logit_ws,
                   unsigned* __restrict__ max_ws) {
  __shared__ __align__(16) _Float16 s_attr[32][EDGE_DIM];
  __shared__ int   s_src[32], s_dst[32];
  __shared__ float s_relt[32];
  __shared__ float s_logit[32][2];

  const int tid = threadIdx.x;
  const int base = blockIdx.x * 32;

  if (tid < 32) {
    int e = base + tid;
    int s = ei[e];
    s_src[tid]  = s;
    s_dst[tid]  = ei[N_EDGES + e];
    s_relt[tid] = last_update[s] - t[e];
  }
  if (tid < 64) s_logit[tid >> 1][tid & 1] = 0.0f;
  __syncthreads();
  {
    int c = tid;
    if (c < 128) {
      float tw = time_w[c], tb = time_b[c];
#pragma unroll
      for (int ee = 0; ee < 32; ++ee)
        s_attr[ee][c] = (_Float16)cosf(s_relt[ee] * tw + tb);
    } else {
      int mc = c - 128;
#pragma unroll
      for (int ee = 0; ee < 32; ++ee)
        s_attr[ee][c] = (_Float16)msg[(size_t)(base + ee) * 128 + mc];
    }
  }
  __syncthreads();

  const int lane = tid & 31;
  const int wave = tid >> 5;
  const int mrow = lane & 15;
  const bool hi = (lane & 16) != 0;
  const int col = wave * 16 + mrow;
  const int er0 = mrow, er1 = 16 + mrow;

  v8f acc0 = {}, acc1 = {};
#pragma unroll
  for (int kt = 0; kt < 8; ++kt) {
    const int akb = kt * 32 + (hi ? 16 : 0);
    const int bkb = kt * 32 + (hi ? 16 : 0);
    const _Float16* wr = weh + (size_t)col * EDGE_DIM + bkb;
    const v16h b  = mk16(*(const v8h*)wr, *(const v8h*)(wr + 8));
    const v16h a0 = mk16(*(const v8h*)&s_attr[er0][akb],
                         *(const v8h*)&s_attr[er0][akb + 8]);
    const v16h a1 = mk16(*(const v8h*)&s_attr[er1][akb],
                         *(const v8h*)&s_attr[er1][akb + 8]);
    acc0 = __builtin_amdgcn_wmma_f32_16x16x32_f16(
        false, a0, false, b, (short)0, acc0, false, false);
    acc1 = __builtin_amdgcn_wmma_f32_16x16x32_f16(
        false, a1, false, b, (short)0, acc1, false, false);
  }

  const int h = col >> 6;
#pragma unroll
  for (int r = 0; r < 8; ++r) {
    int ee = r + (hi ? 8 : 0);
    float kv0 = acc0[r] + k_ws[(size_t)s_src[ee] * HC + col];
    float qv0 = q_ws[(size_t)s_dst[ee] * HC + col];
    atomicAdd(&s_logit[ee][h], kv0 * qv0);
    int e2 = 16 + ee;
    float kv1 = acc1[r] + k_ws[(size_t)s_src[e2] * HC + col];
    float qv1 = q_ws[(size_t)s_dst[e2] * HC + col];
    atomicAdd(&s_logit[e2][h], kv1 * qv1);
  }
  __syncthreads();

  if (tid < 64) {
    int ee = tid >> 1, hh = tid & 1;
    float l = s_logit[ee][hh] * 0.125f;
    logit_ws[(size_t)(base + ee) * 2 + hh] = l;
    atomicMax(&max_ws[(size_t)s_dst[ee] * 2 + hh], enc_f32(l));
  }
}

__global__ void k_softmax_denom(const int* __restrict__ ei,
                                const unsigned* __restrict__ max_ws,
                                float* __restrict__ logit_ws,
                                float* __restrict__ den_ws) {
  int i = blockIdx.x * 256 + threadIdx.x;
  if (i >= N_EDGES * 2) return;
  int e = i >> 1, h = i & 1;
  int dst = ei[N_EDGES + e];
  float m = dec_f32(max_ws[(size_t)dst * 2 + h]);
  float ex = __expf(logit_ws[i] - m);
  logit_ws[i] = ex;
  atomicAdd(&den_ws[(size_t)dst * 2 + h], ex);
}

__global__ __launch_bounds__(256)
void k_edge_aggregate(const float* __restrict__ last_update,
                      const int* __restrict__ ei,
                      const float* __restrict__ t,
                      const float* __restrict__ msg,
                      const float* __restrict__ time_w,
                      const float* __restrict__ time_b,
                      const _Float16* __restrict__ weh,
                      const float* __restrict__ v_ws,
                      const float* __restrict__ ex_ws,
                      const float* __restrict__ den_ws,
                      float* __restrict__ out) {
  __shared__ __align__(16) _Float16 s_attr[32][EDGE_DIM];
  __shared__ int   s_src[32], s_dst[32];
  __shared__ float s_relt[32];
  __shared__ float s_alpha[32][2];

  const int tid = threadIdx.x;
  const int base = blockIdx.x * 32;

  if (tid < 32) {
    int e = base + tid;
    int s = ei[e];
    s_src[tid]  = s;
    s_dst[tid]  = ei[N_EDGES + e];
    s_relt[tid] = last_update[s] - t[e];
  }
  __syncthreads();

  if (tid < 64) {
    int ee = tid >> 1, hh = tid & 1;
    float d = den_ws[(size_t)s_dst[ee] * 2 + hh];
    s_alpha[ee][hh] = ex_ws[(size_t)(base + ee) * 2 + hh] / d;
  }
  {
    int c = tid;
    if (c < 128) {
      float tw = time_w[c], tb = time_b[c];
#pragma unroll
      for (int ee = 0; ee < 32; ++ee)
        s_attr[ee][c] = (_Float16)cosf(s_relt[ee] * tw + tb);
    } else {
      int mc = c - 128;
#pragma unroll
      for (int ee = 0; ee < 32; ++ee)
        s_attr[ee][c] = (_Float16)msg[(size_t)(base + ee) * 128 + mc];
    }
  }
  __syncthreads();

  const int lane = tid & 31;
  const int wave = tid >> 5;
  const int mrow = lane & 15;
  const bool hi = (lane & 16) != 0;
  const int col = wave * 16 + mrow;
  const int er0 = mrow, er1 = 16 + mrow;

  v8f acc0 = {}, acc1 = {};
#pragma unroll
  for (int kt = 0; kt < 8; ++kt) {
    const int akb = kt * 32 + (hi ? 16 : 0);
    const int bkb = kt * 32 + (hi ? 16 : 0);
    const _Float16* wr = weh + (size_t)col * EDGE_DIM + bkb;
    const v16h b  = mk16(*(const v8h*)wr, *(const v8h*)(wr + 8));
    const v16h a0 = mk16(*(const v8h*)&s_attr[er0][akb],
                         *(const v8h*)&s_attr[er0][akb + 8]);
    const v16h a1 = mk16(*(const v8h*)&s_attr[er1][akb],
                         *(const v8h*)&s_attr[er1][akb + 8]);
    acc0 = __builtin_amdgcn_wmma_f32_16x16x32_f16(
        false, a0, false, b, (short)0, acc0, false, false);
    acc1 = __builtin_amdgcn_wmma_f32_16x16x32_f16(
        false, a1, false, b, (short)0, acc1, false, false);
  }

  const int h = col >> 6;
#pragma unroll
  for (int r = 0; r < 8; ++r) {
    int ee = r + (hi ? 8 : 0);
    float vv0 = acc0[r] + v_ws[(size_t)s_src[ee] * HC + col];
    atomicAdd(&out[(size_t)s_dst[ee] * HC + col], vv0 * s_alpha[ee][h]);
    int e2 = 16 + ee;
    float vv1 = acc1[r] + v_ws[(size_t)s_src[e2] * HC + col];
    atomicAdd(&out[(size_t)s_dst[e2] * HC + col], vv1 * s_alpha[e2][h]);
  }
}

// ------------------------------------------------------------------
extern "C" void kernel_launch(void* const* d_in, const int* in_sizes, int n_in,
                              void* d_out, int out_size, void* d_ws, size_t ws_size,
                              hipStream_t stream) {
  (void)in_sizes; (void)n_in; (void)out_size;
  const float* x           = (const float*)d_in[0];
  const float* last_update = (const float*)d_in[1];
  const int*   ei          = (const int*)  d_in[2];
  const float* t           = (const float*)d_in[3];
  const float* msg         = (const float*)d_in[4];
  const float* time_w      = (const float*)d_in[5];
  const float* time_b      = (const float*)d_in[6];
  const float* Wq = (const float*)d_in[7];  const float* bq = (const float*)d_in[8];
  const float* Wk = (const float*)d_in[9];  const float* bk = (const float*)d_in[10];
  const float* Wv = (const float*)d_in[11]; const float* bv = (const float*)d_in[12];
  const float* We = (const float*)d_in[13];
  const float* Ws = (const float*)d_in[14]; const float* bs = (const float*)d_in[15];
  float* out = (float*)d_out;

  // ---- common workspace prefix ----
  _Float16* xh  = (_Float16*)d_ws;                         // N*128
  _Float16* wh  = xh + (size_t)N_NODES * D_IN;             // 4*128*128
  _Float16* weh = wh + (size_t)4 * D_IN * HC;              // 128*256
  float*    q_ws     = (float*)(weh + (size_t)HC * EDGE_DIM);
  float*    k_ws     = q_ws + (size_t)N_NODES * HC;
  float*    v_ws     = k_ws + (size_t)N_NODES * HC;
  float*    logit_ws = v_ws + (size_t)N_NODES * HC;        // E*2

  // CSR-path suffix
  int*      deg    = (int*)(logit_ws + (size_t)N_EDGES * 2);
  int*      off    = deg + N_NODES;
  int*      cursor = off + N_NODES;
  int*      bsum   = cursor + N_NODES;                     // 256
  int*      bofs   = bsum + 256;                           // 256
  int*      csr    = bofs + 256;                           // E
  _Float16* vh     = (_Float16*)(csr + N_EDGES);           // E*128 f16

  const size_t csr_need = (size_t)((char*)(vh + (size_t)N_EDGES * HC) - (char*)d_ws);

  const int NW = D_IN * HC;  // 16384
  k_to_f16<<<(N_NODES * D_IN + 255) / 256, 256, 0, stream>>>(x, xh, N_NODES * D_IN);
  k_to_f16<<<(NW + 255) / 256, 256, 0, stream>>>(Wq, wh + 0 * NW, NW);
  k_to_f16<<<(NW + 255) / 256, 256, 0, stream>>>(Wk, wh + 1 * NW, NW);
  k_to_f16<<<(NW + 255) / 256, 256, 0, stream>>>(Wv, wh + 2 * NW, NW);
  k_to_f16<<<(NW + 255) / 256, 256, 0, stream>>>(Ws, wh + 3 * NW, NW);
  k_to_f16<<<(HC * EDGE_DIM + 255) / 256, 256, 0, stream>>>(We, weh, HC * EDGE_DIM);

  k_node_proj<<<N_NODES / 16, 256, 0, stream>>>(
      xh, wh, bq, bk, bv, bs, q_ws, k_ws, v_ws, out);

  if (ws_size >= csr_need) {
    // ---- primary pipeline: CSR + atomic-free aggregation ----
    const int NB = (N_NODES + 255) / 256;   // 196 blocks (<= 256 for scan2)
    k_zero_int<<<NB, 256, 0, stream>>>(deg, N_NODES);
    k_degree<<<(N_EDGES + 255) / 256, 256, 0, stream>>>(ei, deg);
    k_scan1<<<NB, 256, 0, stream>>>(deg, off, bsum);
    k_scan2<<<1, 256, 0, stream>>>(bsum, bofs, NB);
    k_scan3<<<NB, 256, 0, stream>>>(off, bofs, cursor);
    k_scatter<<<(N_EDGES + 255) / 256, 256, 0, stream>>>(ei, cursor, csr);

    k_edge_compute<<<N_EDGES / 32, 256, 0, stream>>>(
        last_update, ei, t, msg, time_w, time_b, weh,
        q_ws, k_ws, v_ws, logit_ws, vh);
    k_node_agg<<<N_NODES / 8, 256, 0, stream>>>(off, csr, logit_ws, vh, out);
  } else {
    // ---- fallback: atomic softmax + recompute pipeline ----
    unsigned* max_ws = (unsigned*)(logit_ws + (size_t)N_EDGES * 2);
    float*    den_ws = (float*)(max_ws + (size_t)N_NODES * 2);
    k_init<<<(N_NODES * 2 + 255) / 256, 256, 0, stream>>>(max_ws, den_ws);
    k_edge_logits<<<N_EDGES / 32, 256, 0, stream>>>(
        last_update, ei, t, msg, time_w, time_b, weh, q_ws, k_ws, logit_ws, max_ws);
    k_softmax_denom<<<(N_EDGES * 2 + 255) / 256, 256, 0, stream>>>(
        ei, max_ws, logit_ws, den_ws);
    k_edge_aggregate<<<N_EDGES / 32, 256, 0, stream>>>(
        last_update, ei, t, msg, time_w, time_b, weh, v_ws, logit_ws, den_ws, out);
  }
}
